// IrisV4Enhanced_8358006358363
// MI455X (gfx1250) — matrix operand order repaired and checked by reading, not compiled, not run on hardware
//
#include <hip/hip_runtime.h>
#include <hip/hip_bf16.h>
#include <math.h>

// ---------------- problem constants ----------------
constexpr int BB   = 8;      // batch
constexpr int HH   = 8;      // heads
constexpr int SL   = 900;    // sequence (30x30)
constexpr int SPJ  = 928;    // padded j (multiple of 32)
constexpr int SPI  = 912;    // padded i (multiple of 16)
constexpr int DD   = 512;    // d_model
constexpr int DK   = 64;     // head dim
constexpr int FF   = 2048;   // ffn hidden
constexpr int NCc  = 10;     // colors
constexpr float SCALE = 0.125f;  // 1/sqrt(64)

// ---------------- types ----------------
typedef __attribute__((ext_vector_type(16))) __bf16       v16bf;
typedef __attribute__((ext_vector_type(8)))  float        v8f;
typedef __attribute__((ext_vector_type(8)))  unsigned int v8u;
typedef __attribute__((ext_vector_type(4)))  int          v4i;
// address-space qualified element types for the async-LDS builtin
typedef __attribute__((address_space(1))) v4i v4i_g;   // global
typedef __attribute__((address_space(3))) v4i v4i_l;   // LDS

// ---------------- gfx1250 async global->LDS path (guarded) ----------------
#if defined(__has_builtin)
#  if __has_builtin(__builtin_amdgcn_global_load_async_to_lds_b128) && \
      __has_builtin(__builtin_amdgcn_s_wait_asynccnt)
#    define USE_ASYNC_LDS 1
#  endif
#endif

__device__ __forceinline__ void copy16_g2l(const unsigned short* g, unsigned short* l) {
#ifdef USE_ASYNC_LDS
  __builtin_amdgcn_global_load_async_to_lds_b128((v4i_g*)g, (v4i_l*)l, 0, 0);
#else
  *reinterpret_cast<uint4*>(l) = *reinterpret_cast<const uint4*>(g);
#endif
}

__device__ __forceinline__ void wait_async() {
#ifdef USE_ASYNC_LDS
  __builtin_amdgcn_s_wait_asynccnt(0);
#endif
}

__device__ __forceinline__ unsigned short f2bf_bits(float x) {
  unsigned int u = __float_as_uint(x);
  unsigned int r = u + 0x7FFFu + ((u >> 16) & 1u);
  return (unsigned short)(r >> 16);
}

// A fragment: 16x32 bf16 tile from row-major [*, ld] at (row0, k0).
// lanes 0-15: M=lane, regs0-3 K=0..7, regs4-7 K=16..23; lanes 16-31: +8 on K base.
__device__ __forceinline__ v16bf frag_a(const unsigned short* A, int ld,
                                        int row0, int k0, int lane) {
  int m  = lane & 15;
  int kb = (lane >> 4) << 3;
  const uint4* p0 = reinterpret_cast<const uint4*>(A + (size_t)(row0 + m) * ld + k0 + kb);
  const uint4* p1 = reinterpret_cast<const uint4*>(A + (size_t)(row0 + m) * ld + k0 + 16 + kb);
  uint4 a = *p0, b = *p1;
  v8u r = {a.x, a.y, a.z, a.w, b.x, b.y, b.z, b.w};
  return __builtin_bit_cast(v16bf, r);
}

// B fragment: 32x16 bf16 tile from K-transposed storage Bt[N][K] at (n0, k0).
__device__ __forceinline__ v16bf frag_b(const unsigned short* Bt, int ld,
                                        int n0, int k0, int lane) {
  int n  = lane & 15;
  int kb = (lane >> 4) << 3;
  const uint4* p0 = reinterpret_cast<const uint4*>(Bt + (size_t)(n0 + n) * ld + k0 + kb);
  const uint4* p1 = reinterpret_cast<const uint4*>(Bt + (size_t)(n0 + n) * ld + k0 + 16 + kb);
  uint4 a = *p0, b = *p1;
  v8u r = {a.x, a.y, a.z, a.w, b.x, b.y, b.z, b.w};
  return __builtin_bit_cast(v16bf, r);
}

__device__ __forceinline__ float gelu_exact(float x) {
  return 0.5f * x * (1.0f + erff(x * 0.70710678118654752f));
}

#define WMMA_BF16(a, b, c) \
  __builtin_amdgcn_wmma_f32_16x16x32_bf16(false, (a), false, (b), (short)0, (c), false, false)

// ---------------- utility kernels ----------------
__global__ void k_zero_u32(unsigned int* p, size_t n) {
  size_t i = (size_t)blockIdx.x * blockDim.x + threadIdx.x;
  if (i < n) p[i] = 0u;
}

__global__ void k_f32_to_bf16(const float* X, unsigned short* Y, size_t n) {
  size_t i = (size_t)blockIdx.x * blockDim.x + threadIdx.x;
  if (i < n) Y[i] = f2bf_bits(X[i]);
}

// W [K][N] fp32 -> Wt [N][K] bf16
__global__ void k_packT(const float* W, unsigned short* Wt, int K, int N) {
  size_t i = (size_t)blockIdx.x * blockDim.x + threadIdx.x;
  if (i >= (size_t)K * N) return;
  int n = (int)(i % N), k = (int)(i / N);
  Wt[(size_t)n * K + k] = f2bf_bits(W[i]);
}

// qf/kf/vf fp32 [B*S][512] -> padded per-head bf16 Q,K [bh][SPJ][64], Vt [bh][64][SPJ]
__global__ void k_pack_qkv(const float* qf, const float* kf, const float* vf,
                           unsigned short* Qb, unsigned short* Kb, unsigned short* Vt) {
  size_t i = (size_t)blockIdx.x * blockDim.x + threadIdx.x;
  if (i >= (size_t)BB * SL * DD) return;
  int d = (int)(i % DD);
  size_t bs = i / DD;
  int s = (int)(bs % SL);
  int b = (int)(bs / SL);
  int h = d >> 6, dk = d & 63;
  size_t bh = (size_t)(b * HH + h);
  Qb[(bh * SPJ + s) * DK + dk] = f2bf_bits(qf[i]);
  Kb[(bh * SPJ + s) * DK + dk] = f2bf_bits(kf[i]);
  Vt[(bh * DK + dk) * SPJ + s] = f2bf_bits(vf[i]);
}

// ---------------- LDS-staged double-buffered bf16 WMMA GEMM ----------------
// C = act(A @ Bt^T + bias) + resid ; A bf16 [M][K], Bt bf16 [N][K]
// block 256 = 8 waves; each wave 32(M) x 64(N); block tile 256 x 64.
// Tiles staged into LDS with async global->LDS copies (ASYNCcnt), double buffered.
constexpr int GT_M = 256;
constexpr int GT_N = 64;
__global__ void __launch_bounds__(256)
k_gemm(const unsigned short* __restrict__ A,
       const unsigned short* __restrict__ Bt,
       const float* __restrict__ bias,
       const float* __restrict__ resid,
       float* __restrict__ Cf, unsigned short* __restrict__ Cbf,
       int M, int N, int K, int act) {
  __shared__ __align__(16) unsigned short lA[2][GT_M * 32];  // 16KB x2
  __shared__ __align__(16) unsigned short lB[2][GT_N * 32];  // 4KB x2
  const int tid  = threadIdx.x;
  const int lane = tid & 31;
  const int w    = tid >> 5;
  const int R0   = blockIdx.y * GT_M;
  const int n0   = blockIdx.x * GT_N;
  const int r0   = w * 32;  // wave-local row base within the block tile

  auto stage = [&](int k0, int buf) {
    // A tile: GT_M x 32 elems = 1024 16B-chunks, 4 per thread
#pragma unroll
    for (int i = 0; i < 4; i++) {
      int chunk = tid + 256 * i;
      int row = chunk >> 2, cg = (chunk & 3) << 3;
      int grow = R0 + row; if (grow >= M) grow = M - 1;   // clamp (stores are guarded)
      copy16_g2l(A + (size_t)grow * K + k0 + cg, &lA[buf][row * 32 + cg]);
    }
    // B tile: GT_N x 32 elems = 256 16B-chunks, 1 per thread
    {
      int row = tid >> 2, cg = (tid & 3) << 3;
      copy16_g2l(Bt + (size_t)(n0 + row) * K + k0 + cg, &lB[buf][row * 32 + cg]);
    }
  };

  v8f z = {0.f, 0.f, 0.f, 0.f, 0.f, 0.f, 0.f, 0.f};
  v8f acc[2][4] = {{z, z, z, z}, {z, z, z, z}};

  stage(0, 0);
  const int nk = K / 32;
  for (int kt = 0; kt < nk; kt++) {
    wait_async();
    __syncthreads();                       // staged tile visible block-wide
    int buf = kt & 1;
    if (kt + 1 < nk) stage((kt + 1) * 32, buf ^ 1);  // fetch ahead into other buffer
    v16bf a0 = frag_a(lA[buf], 32, r0, 0, lane);
    v16bf a1 = frag_a(lA[buf], 32, r0 + 16, 0, lane);
#pragma unroll
    for (int t = 0; t < 4; t++) {
      v16bf bfr = frag_b(lB[buf], 32, t * 16, 0, lane);
      acc[0][t] = WMMA_BF16(a0, bfr, acc[0][t]);
      acc[1][t] = WMMA_BF16(a1, bfr, acc[1][t]);
    }
    __syncthreads();                       // done reading buf before it is refilled
  }

  const int half8 = (lane >> 4) * 8;
  const int nl    = lane & 15;
#pragma unroll
  for (int mi = 0; mi < 2; mi++) {
#pragma unroll
    for (int t = 0; t < 4; t++) {
      int col = n0 + t * 16 + nl;
      float bv = bias ? bias[col] : 0.f;
#pragma unroll
      for (int e = 0; e < 8; e++) {
        int row = R0 + r0 + mi * 16 + half8 + e;
        if (row < M) {
          float v = acc[mi][t][e] + bv;
          if (act == 1) v = gelu_exact(v);
          size_t o = (size_t)row * N + col;
          if (resid) v += resid[o];
          if (Cf)  Cf[o]  = v;
          if (Cbf) Cbf[o] = f2bf_bits(v);
        }
      }
    }
  }
}

// ---------------- attention pass A: per-row softmax stats ----------------
// grid (SPI/16, B), block 256 (8 waves = 8 heads)
__global__ void __launch_bounds__(256)
k_attn_stats(const unsigned short* __restrict__ Qb_,
             const unsigned short* __restrict__ Kb_,
             const int* __restrict__ cidx,
             const float* __restrict__ cbias,
             float* stMB, float* stSB, float* stMU, float* stSU) {
  __shared__ float s_bias[NCc * NCc * HH];
  __shared__ int   s_ci[16];
  __shared__ int   s_cj[SPJ];
  int b  = blockIdx.y;
  int i0 = blockIdx.x * 16;
  int lane = threadIdx.x & 31;
  int h    = threadIdx.x >> 5;

  for (int i = threadIdx.x; i < NCc * NCc * HH; i += blockDim.x) s_bias[i] = cbias[i];
  for (int j = threadIdx.x; j < SPJ; j += blockDim.x)
    s_cj[j] = (j < SL) ? cidx[b * SL + j] : 0;
  if (threadIdx.x < 16) {
    int ii = i0 + threadIdx.x;
    s_ci[threadIdx.x] = (ii < SL) ? cidx[b * SL + ii] : 0;
  }
  __syncthreads();

  const unsigned short* Qb = Qb_ + (size_t)(b * HH + h) * SPJ * DK;
  const unsigned short* Kb = Kb_ + (size_t)(b * HH + h) * SPJ * DK;
  int half8 = (lane >> 4) * 8;

  // loop-invariant Q fragments
  v16bf qa0 = frag_a(Qb, DK, i0, 0, lane);
  v16bf qa1 = frag_a(Qb, DK, i0, 32, lane);

  float mB[8], sB[8], mU[8], sU[8];
#pragma unroll
  for (int e = 0; e < 8; e++) { mB[e] = -3e38f; sB[e] = 0.f; mU[e] = -3e38f; sU[e] = 0.f; }

  const int NT = SPJ / 16;
  // software-pipelined K fragments: fetch tile jt+1 while computing tile jt
  v16bf kb0 = frag_b(Kb, DK, 0, 0, lane);
  v16bf kb1 = frag_b(Kb, DK, 0, 32, lane);
  for (int jt = 0; jt < NT; jt++) {
    int j0 = jt * 16;
    v16bf nb0, nb1;
    if (jt + 1 < NT) {
      __builtin_prefetch((const void*)(Kb + (size_t)(j0 + 32) * DK), 0, 1);
      nb0 = frag_b(Kb, DK, j0 + 16, 0, lane);
      nb1 = frag_b(Kb, DK, j0 + 16, 32, lane);
    }
    v8f c = {0.f, 0.f, 0.f, 0.f, 0.f, 0.f, 0.f, 0.f};
    c = WMMA_BF16(qa0, kb0, c);
    c = WMMA_BF16(qa1, kb1, c);

    int j = j0 + (lane & 15);
    bool jv = (j < SL);
    int cj = s_cj[j];
#pragma unroll
    for (int e = 0; e < 8; e++) {
      float sc = jv ? c[e] * SCALE : -1e30f;
      if (sc > mU[e]) { sU[e] = sU[e] * __expf(mU[e] - sc) + 1.f; mU[e] = sc; }
      else            { sU[e] += __expf(sc - mU[e]); }
      float sb = jv ? (sc + s_bias[(s_ci[half8 + e] * NCc + cj) * HH + h]) : -1e30f;
      if (sb > mB[e]) { sB[e] = sB[e] * __expf(mB[e] - sb) + 1.f; mB[e] = sb; }
      else            { sB[e] += __expf(sb - mB[e]); }
    }
    if (jt + 1 < NT) { kb0 = nb0; kb1 = nb1; }
  }

  // reduce across the 16 lanes that share each row
  for (int off = 1; off < 16; off <<= 1) {
#pragma unroll
    for (int e = 0; e < 8; e++) {
      float mo = __shfl_xor(mU[e], off, 32), so = __shfl_xor(sU[e], off, 32);
      float mn = fmaxf(mU[e], mo);
      sU[e] = sU[e] * __expf(mU[e] - mn) + so * __expf(mo - mn);
      mU[e] = mn;
      mo = __shfl_xor(mB[e], off, 32); so = __shfl_xor(sB[e], off, 32);
      mn = fmaxf(mB[e], mo);
      sB[e] = sB[e] * __expf(mB[e] - mn) + so * __expf(mo - mn);
      mB[e] = mn;
    }
  }
  if ((lane & 15) == 0) {
#pragma unroll
    for (int e = 0; e < 8; e++) {
      size_t o = (size_t)(b * HH + h) * SPI + i0 + half8 + e;
      stMB[o] = mB[e]; stSB[o] = sB[e]; stMU[o] = mU[e]; stSU[o] = sU[e];
    }
  }
}

// ---------------- attention pass B: O = softmax(S+bias) @ V, and aw output ----
// grid (SPI/16, B), block 256 (8 waves = 8 heads)
__global__ void __launch_bounds__(256)
k_attn_out(const unsigned short* __restrict__ Qb_,
           const unsigned short* __restrict__ Kb_,
           const unsigned short* __restrict__ Vt_,
           const int* __restrict__ cidx,
           const float* __restrict__ cbias,
           const float* stMB, const float* stSB,
           const float* stMU, const float* stSU,
           float* __restrict__ aw,
           unsigned short* __restrict__ attnbf) {
  __shared__ float s_bias[NCc * NCc * HH];
  __shared__ int   s_ci[16];
  __shared__ int   s_cj[SPJ];
  __shared__ float s_aw[16 * 32];
  __shared__ __align__(16) unsigned short s_p[HH * 16 * 32];

  int b  = blockIdx.y;
  int i0 = blockIdx.x * 16;
  int lane = threadIdx.x & 31;
  int h    = threadIdx.x >> 5;

  for (int i = threadIdx.x; i < NCc * NCc * HH; i += blockDim.x) s_bias[i] = cbias[i];
  for (int j = threadIdx.x; j < SPJ; j += blockDim.x)
    s_cj[j] = (j < SL) ? cidx[b * SL + j] : 0;
  if (threadIdx.x < 16) {
    int ii = i0 + threadIdx.x;
    s_ci[threadIdx.x] = (ii < SL) ? cidx[b * SL + ii] : 0;
  }
  __syncthreads();

  const unsigned short* Qb = Qb_ + (size_t)(b * HH + h) * SPJ * DK;
  const unsigned short* Kb = Kb_ + (size_t)(b * HH + h) * SPJ * DK;
  const unsigned short* Vt = Vt_ + (size_t)(b * HH + h) * DK * SPJ;
  unsigned short* sp = s_p + h * 16 * 32;
  int half8 = (lane >> 4) * 8;

  float mB[8], rB[8], mU[8], rU[8];
#pragma unroll
  for (int e = 0; e < 8; e++) {
    size_t o = (size_t)(b * HH + h) * SPI + i0 + half8 + e;
    mB[e] = stMB[o]; rB[e] = 1.f / stSB[o];
    mU[e] = stMU[o]; rU[e] = 1.f / stSU[o];
  }

  // loop-invariant Q fragments
  v16bf qa0 = frag_a(Qb, DK, i0, 0, lane);
  v16bf qa1 = frag_a(Qb, DK, i0, 32, lane);

  v8f z = {0.f, 0.f, 0.f, 0.f, 0.f, 0.f, 0.f, 0.f};
  v8f oacc[4] = {z, z, z, z};

  for (int jc = 0; jc < SPJ / 32; jc++) {
    int jbase = jc * 32;
    // preload this chunk's K and V fragments (independent global loads, in
    // flight while we zero s_aw / run VALU work)
    v16bf kb[4], vb[4];
    kb[0] = frag_b(Kb, DK, jbase,      0,  lane);
    kb[1] = frag_b(Kb, DK, jbase,      32, lane);
    kb[2] = frag_b(Kb, DK, jbase + 16, 0,  lane);
    kb[3] = frag_b(Kb, DK, jbase + 16, 32, lane);
#pragma unroll
    for (int t = 0; t < 4; t++) vb[t] = frag_b(Vt, SPJ, t * 16, jbase, lane);

    for (int i = threadIdx.x; i < 512; i += blockDim.x) s_aw[i] = 0.f;
    __syncthreads();

#pragma unroll
    for (int t = 0; t < 2; t++) {
      int j0 = jbase + t * 16;
      v8f c = z;
      c = WMMA_BF16(qa0, kb[2 * t], c);
      c = WMMA_BF16(qa1, kb[2 * t + 1], c);
      int n = lane & 15;
      int j = j0 + n;
      bool jv = (j < SL);
      int cj = s_cj[j];
#pragma unroll
      for (int e = 0; e < 8; e++) {
        float pu = 0.f, pb = 0.f;
        if (jv) {
          float sc = c[e] * SCALE;
          pu = __expf(sc - mU[e]) * rU[e];
          float bv = s_bias[(s_ci[half8 + e] * NCc + cj) * HH + h];
          pb = __expf(sc + bv - mB[e]) * rB[e];
        }
        atomicAdd(&s_aw[(half8 + e) * 32 + t * 16 + n], pu * (1.f / HH));
        sp[(half8 + e) * 32 + t * 16 + n] = f2bf_bits(pb);
      }
    }
    __syncthreads();

    // O += P (16x32) @ V (32x64)
    v16bf pA = frag_a(sp, 32, 0, 0, lane);
#pragma unroll
    for (int t = 0; t < 4; t++)
      oacc[t] = WMMA_BF16(pA, vb[t], oacc[t]);

    // write head-averaged unbiased softmax tile
    for (int idx = threadIdx.x; idx < 512; idx += blockDim.x) {
      int rl = idx >> 5, jj = idx & 31;
      int gi = i0 + rl, gj = jbase + jj;
      if (gi < SL && gj < SL)
        aw[((size_t)b * SL + gi) * SL + gj] = s_aw[idx];
    }
    __syncthreads();
  }

  // write O as bf16 [B*S][512] with col = h*64 + dk
#pragma unroll
  for (int t = 0; t < 4; t++) {
    int col = h * DK + t * 16 + (lane & 15);
#pragma unroll
    for (int e = 0; e < 8; e++) {
      int gi = i0 + half8 + e;
      if (gi < SL)
        attnbf[((size_t)b * SL + gi) * DD + col] = f2bf_bits(oacc[t][e]);
    }
  }
}

// ---------------- layernorm (one wave per row, D=512) ----------------
__global__ void k_layernorm(const float* __restrict__ X, const float* __restrict__ g,
                            const float* __restrict__ bt, float* __restrict__ Y,
                            unsigned short* __restrict__ Ybf, int M) {
  int row  = blockIdx.x * 8 + (threadIdx.x >> 5);
  int lane = threadIdx.x & 31;
  if (row >= M) return;
  const float* x = X + (size_t)row * DD;
  float v[16];
  float s = 0.f, s2 = 0.f;
#pragma unroll
  for (int t = 0; t < 16; t++) {
    float xv = x[lane + 32 * t];
    v[t] = xv; s += xv; s2 += xv * xv;
  }
  for (int off = 1; off < 32; off <<= 1) {
    s  += __shfl_xor(s, off, 32);
    s2 += __shfl_xor(s2, off, 32);
  }
  float mean = s * (1.f / DD);
  float var  = s2 * (1.f / DD) - mean * mean;
  float inv  = rsqrtf(var + 1e-5f);
#pragma unroll
  for (int t = 0; t < 16; t++) {
    int d = lane + 32 * t;
    float yv = (v[t] - mean) * inv * g[d] + bt[d];
    Y[(size_t)row * DD + d] = yv;
    if (Ybf) Ybf[(size_t)row * DD + d] = f2bf_bits(yv);
  }
}

// gf[b][d] = mean_s out[b][s][d]
__global__ void k_pool(const float* __restrict__ X, float* __restrict__ gf) {
  int b = blockIdx.x;
  for (int d = threadIdx.x; d < DD; d += blockDim.x) {
    float s = 0.f;
    for (int i = 0; i < SL; i++) s += X[((size_t)b * SL + i) * DD + d];
    gf[b * DD + d] = s * (1.f / SL);
  }
}

// tiny fp32 GEMM for the M=8 heads
__global__ void k_small_gemm(const float* __restrict__ A, const float* __restrict__ W,
                             const float* __restrict__ bias, float* __restrict__ C,
                             int M, int K, int N, int act) {
  int idx = blockIdx.x * blockDim.x + threadIdx.x;
  if (idx >= M * N) return;
  int m = idx / N, n = idx % N;
  float s = bias ? bias[n] : 0.f;
  for (int k = 0; k < K; k++) s += A[(size_t)m * K + k] * W[(size_t)k * N + n];
  if (act == 1) s = gelu_exact(s);
  C[idx] = s;
}

__global__ void k_softmax_rows(const float* __restrict__ X, float* __restrict__ Y,
                               int M, int N) {
  int r = blockIdx.x * blockDim.x + threadIdx.x;
  if (r >= M) return;
  float mx = -3e38f;
  for (int j = 0; j < N; j++) mx = fmaxf(mx, X[(size_t)r * N + j]);
  float s = 0.f;
  for (int j = 0; j < N; j++) s += __expf(X[(size_t)r * N + j] - mx);
  float inv = 1.f / s;
  for (int j = 0; j < N; j++) Y[(size_t)r * N + j] = __expf(X[(size_t)r * N + j] - mx) * inv;
}

__global__ void k_concat(const float* gf, const float* hue, const float* sat,
                         const float* bri, const float* harm, float* comb) {
  int b = blockIdx.x;
  for (int i = threadIdx.x; i < 576; i += blockDim.x) {
    float v;
    if (i < 512)      v = gf[b * 512 + i];
    else if (i < 528) v = hue[b * 16 + (i - 512)];
    else if (i < 536) v = sat[b * 8 + (i - 528)];
    else if (i < 544) v = bri[b * 8 + (i - 536)];
    else              v = harm[b * 32 + (i - 544)];
    comb[b * 576 + i] = v;
  }
}

// ---------------- launch ----------------
extern "C" void kernel_launch(void* const* d_in, const int* in_sizes, int n_in,
                              void* d_out, int out_size, void* d_ws, size_t ws_size,
                              hipStream_t stream) {
  (void)in_sizes; (void)n_in; (void)out_size; (void)ws_size;

  const float* x       = (const float*)d_in[0];   // [8,30,30,512] == [7200,512]
  const int*   cidx    = (const int*)d_in[1];     // [8,900]
  const float* wq_w    = (const float*)d_in[2];
  const float* wq_b    = (const float*)d_in[3];
  const float* wk_w    = (const float*)d_in[4];
  const float* wk_b    = (const float*)d_in[5];
  const float* wv_w    = (const float*)d_in[6];
  const float* wv_b    = (const float*)d_in[7];
  const float* wo_w    = (const float*)d_in[8];
  const float* wo_b    = (const float*)d_in[9];
  const float* cbias   = (const float*)d_in[10];  // [10,10,8]
  const float* aln_g   = (const float*)d_in[11];
  const float* aln_b   = (const float*)d_in[12];
  const float* hue0w   = (const float*)d_in[13];
  const float* hue0b   = (const float*)d_in[14];
  const float* hue1w   = (const float*)d_in[15];
  const float* hue1b   = (const float*)d_in[16];
  const float* sat0w   = (const float*)d_in[17];
  const float* sat0b   = (const float*)d_in[18];
  const float* sat1w   = (const float*)d_in[19];
  const float* sat1b   = (const float*)d_in[20];
  const float* bri0w   = (const float*)d_in[21];
  const float* bri0b   = (const float*)d_in[22];
  const float* bri1w   = (const float*)d_in[23];
  const float* bri1b   = (const float*)d_in[24];
  const float* harm0w  = (const float*)d_in[25];
  const float* harm0b  = (const float*)d_in[26];
  const float* harm1w  = (const float*)d_in[27];
  const float* harm1b  = (const float*)d_in[28];
  const float* ctp0w   = (const float*)d_in[29];  // [576,512]
  const float* ctp0b   = (const float*)d_in[30];
  const float* ctp1w   = (const float*)d_in[31];  // [512,100]
  const float* ctp1b   = (const float*)d_in[32];
  const float* ff0w    = (const float*)d_in[33];
  const float* ff0b    = (const float*)d_in[34];
  const float* ff1w    = (const float*)d_in[35];
  const float* ff1b    = (const float*)d_in[36];
  const float* fln_g   = (const float*)d_in[37];
  const float* fln_b   = (const float*)d_in[38];

  // d_out layout (return-order concatenation)
  float* out_y    = (float*)d_out;                 // 3,686,400
  float* out_aw   = out_y + 3686400;               // 6,480,000
  float* out_ct   = out_aw + (size_t)BB * SL * SL; // 800
  float* out_hue  = out_ct + BB * NCc * NCc;       // 128
  float* out_sat  = out_hue + BB * 16;             // 64
  float* out_bri  = out_sat + BB * 8;              // 64
  float* out_harm = out_bri + BB * 8;              // 256

  // workspace carve-out
  char* cur = (char*)d_ws;
  auto alloc = [&](size_t bytes) -> void* {
    void* p = (void*)cur;
    cur += (bytes + 255) & ~(size_t)255;
    return p;
  };
  const size_t MROWS = (size_t)BB * SL;          // 7200
  unsigned short* xs_bf  = (unsigned short*)alloc(MROWS * DD * 2);
  unsigned short* wqT    = (unsigned short*)alloc((size_t)DD * DD * 2);
  unsigned short* wkT    = (unsigned short*)alloc((size_t)DD * DD * 2);
  unsigned short* wvT    = (unsigned short*)alloc((size_t)DD * DD * 2);
  unsigned short* woT    = (unsigned short*)alloc((size_t)DD * DD * 2);
  unsigned short* ff0T   = (unsigned short*)alloc((size_t)FF * DD * 2);  // [2048][512]
  unsigned short* ff1T   = (unsigned short*)alloc((size_t)DD * FF * 2);  // [512][2048]
  float* qf = (float*)alloc(MROWS * DD * 4);
  float* kf = (float*)alloc(MROWS * DD * 4);
  float* vf = (float*)alloc(MROWS * DD * 4);
  unsigned short* Qbf = (unsigned short*)alloc((size_t)BB * HH * SPJ * DK * 2);
  unsigned short* Kbf = (unsigned short*)alloc((size_t)BB * HH * SPJ * DK * 2);
  unsigned short* Vt  = (unsigned short*)alloc((size_t)BB * HH * DK * SPJ * 2);
  float* stMB = (float*)alloc((size_t)BB * HH * SPI * 4);
  float* stSB = (float*)alloc((size_t)BB * HH * SPI * 4);
  float* stMU = (float*)alloc((size_t)BB * HH * SPI * 4);
  float* stSU = (float*)alloc((size_t)BB * HH * SPI * 4);
  float* outf           = (float*)alloc(MROWS * DD * 4);
  unsigned short* outbf = (unsigned short*)alloc(MROWS * DD * 2);
  float* gf     = (float*)alloc(BB * DD * 4);
  float* hueh   = (float*)alloc(BB * 256 * 4);
  float* huev   = (float*)alloc(BB * 16 * 4);
  float* sath   = (float*)alloc(BB * 256 * 4);
  float* satv   = (float*)alloc(BB * 8 * 4);
  float* brih   = (float*)alloc(BB * 256 * 4);
  float* briv   = (float*)alloc(BB * 8 * 4);
  float* harmh  = (float*)alloc(BB * 512 * 4);
  float* harmp  = (float*)alloc(BB * 32 * 4);
  float* comb   = (float*)alloc(BB * 576 * 4);
  float* ctph   = (float*)alloc(BB * 512 * 4);
  float* ctpp   = (float*)alloc(BB * 100 * 4);
  // aliased regions (lifetimes disjoint): q/k/v fp32 are dead after k_pack_qkv
  float*          attnpre = qf;                    // [7200][512] fp32
  unsigned short* attnbf  = (unsigned short*)kf;   // [7200][512] bf16
  unsigned short* hf_bf   = (unsigned short*)qf;   // [7200][2048] bf16 (spans qf+kf)
  float*          ffpre   = vf;                    // [7200][512] fp32

  const int T = 256;
  // 1) pack activations + weights to bf16 (weights K-transposed for B-fragments)
  {
    size_t n = MROWS * DD;
    k_f32_to_bf16<<<dim3((unsigned)((n + T - 1) / T)), T, 0, stream>>>(x, xs_bf, n);
    k_packT<<<dim3((DD * DD + T - 1) / T), T, 0, stream>>>(wq_w, wqT, DD, DD);
    k_packT<<<dim3((DD * DD + T - 1) / T), T, 0, stream>>>(wk_w, wkT, DD, DD);
    k_packT<<<dim3((DD * DD + T - 1) / T), T, 0, stream>>>(wv_w, wvT, DD, DD);
    k_packT<<<dim3((DD * DD + T - 1) / T), T, 0, stream>>>(wo_w, woT, DD, DD);
    k_packT<<<dim3((DD * FF + T - 1) / T), T, 0, stream>>>(ff0w, ff0T, DD, FF);
    k_packT<<<dim3((FF * DD + T - 1) / T), T, 0, stream>>>(ff1w, ff1T, FF, DD);
  }

  // 2) QKV projections (bf16 WMMA, LDS double-buffered, fp32 out)
  dim3 gQ(DD / GT_N, (unsigned)((MROWS + GT_M - 1) / GT_M));
  k_gemm<<<gQ, T, 0, stream>>>(xs_bf, wqT, wq_b, nullptr, qf, nullptr, (int)MROWS, DD, DD, 0);
  k_gemm<<<gQ, T, 0, stream>>>(xs_bf, wkT, wk_b, nullptr, kf, nullptr, (int)MROWS, DD, DD, 0);
  k_gemm<<<gQ, T, 0, stream>>>(xs_bf, wvT, wv_b, nullptr, vf, nullptr, (int)MROWS, DD, DD, 0);

  // 3) pad-zero + pack per-head bf16 Q/K/Vt
  {
    size_t nw = ((size_t)BB * HH * SPJ * DK * 2) / 4;
    k_zero_u32<<<dim3((unsigned)((nw + T - 1) / T)), T, 0, stream>>>((unsigned int*)Qbf, nw);
    k_zero_u32<<<dim3((unsigned)((nw + T - 1) / T)), T, 0, stream>>>((unsigned int*)Kbf, nw);
    k_zero_u32<<<dim3((unsigned)((nw + T - 1) / T)), T, 0, stream>>>((unsigned int*)Vt, nw);
    size_t n = MROWS * DD;
    k_pack_qkv<<<dim3((unsigned)((n + T - 1) / T)), T, 0, stream>>>(qf, kf, vf, Qbf, Kbf, Vt);
  }

  // 4) flash attention: stats pass + output pass (8 waves = 8 heads per block)
  dim3 gA(SPI / 16, BB);
  k_attn_stats<<<gA, T, 0, stream>>>(Qbf, Kbf, cidx, cbias, stMB, stSB, stMU, stSU);
  k_attn_out<<<gA, T, 0, stream>>>(Qbf, Kbf, Vt, cidx, cbias,
                                   stMB, stSB, stMU, stSU, out_aw, attnbf);

  // 5) out-proj + residual, then LN -> outf / outbf
  k_gemm<<<gQ, T, 0, stream>>>(attnbf, woT, wo_b, x, attnpre, nullptr, (int)MROWS, DD, DD, 0);
  k_layernorm<<<dim3((unsigned)(MROWS / 8)), T, 0, stream>>>(attnpre, aln_g, aln_b, outf, outbf, (int)MROWS);

  // 6) global pool + color-space heads (tiny fp32)
  k_pool<<<dim3(BB), T, 0, stream>>>(outf, gf);
  k_small_gemm<<<dim3((BB * 256 + T - 1) / T), T, 0, stream>>>(gf, hue0w, hue0b, hueh, BB, 512, 256, 1);
  k_small_gemm<<<dim3(1), T, 0, stream>>>(hueh, hue1w, hue1b, huev, BB, 256, 16, 0);
  k_small_gemm<<<dim3((BB * 256 + T - 1) / T), T, 0, stream>>>(gf, sat0w, sat0b, sath, BB, 512, 256, 1);
  k_small_gemm<<<dim3(1), T, 0, stream>>>(sath, sat1w, sat1b, satv, BB, 256, 8, 0);
  k_small_gemm<<<dim3((BB * 256 + T - 1) / T), T, 0, stream>>>(gf, bri0w, bri0b, brih, BB, 512, 256, 1);
  k_small_gemm<<<dim3(1), T, 0, stream>>>(brih, bri1w, bri1b, briv, BB, 256, 8, 0);
  k_small_gemm<<<dim3((BB * 512 + T - 1) / T), T, 0, stream>>>(gf, harm0w, harm0b, harmh, BB, 512, 512, 1);
  k_small_gemm<<<dim3(1), T, 0, stream>>>(harmh, harm1w, harm1b, harmp, BB, 512, 32, 0);
  k_softmax_rows<<<dim3(1), 128, 0, stream>>>(harmp, out_harm, BB, 32);
  k_concat<<<dim3(BB), T, 0, stream>>>(gf, huev, satv, briv, out_harm, comb);
  k_small_gemm<<<dim3((BB * 512 + T - 1) / T), T, 0, stream>>>(comb, ctp0w, ctp0b, ctph, BB, 576, 512, 1);
  k_small_gemm<<<dim3((BB * 100 + T - 1) / T), T, 0, stream>>>(ctph, ctp1w, ctp1b, ctpp, BB, 512, 100, 0);
  k_softmax_rows<<<dim3(1), 128, 0, stream>>>(ctpp, out_ct, BB * NCc, NCc);
  k_softmax_rows<<<dim3(1), 128, 0, stream>>>(huev, out_hue, BB, 16);
  k_softmax_rows<<<dim3(1), 128, 0, stream>>>(satv, out_sat, BB, 8);
  k_softmax_rows<<<dim3(1), 128, 0, stream>>>(briv, out_bri, BB, 8);

  // 7) FFN: h = gelu(out @ ff0 + b0) (bf16), y_pre = h @ ff1 + b1 + out, LN -> y
  dim3 gF1(FF / GT_N, (unsigned)((MROWS + GT_M - 1) / GT_M));
  k_gemm<<<gF1, T, 0, stream>>>(outbf, ff0T, ff0b, nullptr, nullptr, hf_bf, (int)MROWS, FF, DD, 1);
  k_gemm<<<gQ, T, 0, stream>>>(hf_bf, ff1T, ff1b, outf, ffpre, nullptr, (int)MROWS, DD, FF, 0);
  k_layernorm<<<dim3((unsigned)(MROWS / 8)), T, 0, stream>>>(ffpre, fln_g, fln_b, out_y, nullptr, (int)MROWS);
}
